// CrossScaleAttentionMem_24833500905604
// MI455X (gfx1250) — matrix operand order repaired and checked by reading, not compiled, not run on hardware
//
#include <hip/hip_runtime.h>
#include <hip/hip_bf16.h>
#include <math.h>

// ---------------- problem constants ----------------
namespace csa {
constexpr int B      = 2;
constexpr int NT     = 9216;          // tokens (96*96)
constexpr int TT     = 2304;          // downsampled tokens (48*48)
constexpr int D      = 192;           // model dim
constexpr int HEADS  = 6;
constexpr int HD     = 32;            // head dim
constexpr int MEM    = 8;
constexpr int TM     = TT + MEM;      // 2312 keys
constexpr int TMP    = 2336;          // padded keys (73*32)
constexpr int MLPH   = 384;
constexpr int BIN    = TT / MEM;      // 288
constexpr float SCALE = 0.17677669529663687f; // 32^-0.5
constexpr float EMA   = 0.99f;
}

typedef _Float16 v16h __attribute__((ext_vector_type(16)));
typedef _Float16 half8 __attribute__((ext_vector_type(8)));
typedef float    v8f  __attribute__((ext_vector_type(8)));

static __device__ __forceinline__ v16h cat8(half8 lo, half8 hi) {
    return __builtin_shufflevector(lo, hi, 0,1,2,3,4,5,6,7,8,9,10,11,12,13,14,15);
}

// ---------------- weight prep: W[K,N] (f32) -> Wt[N,K] (f16), optional scale ----
__global__ void wprep_kernel(const float* __restrict__ W, _Float16* __restrict__ Wt,
                             int K, int N, float scale) {
    int idx = blockIdx.x * 256 + threadIdx.x;
    if (idx >= K * N) return;
    int k = idx / N, n = idx % N;
    Wt[(size_t)n * K + k] = (_Float16)(scale * W[idx]);
}

__global__ void sbias_kernel(const float* __restrict__ b, float* __restrict__ out, float scale) {
    int i = threadIdx.x;
    if (i < csa::D) out[i] = scale * b[i];
}

// ---------------- 2x2 average-pool downsample ----------------
__global__ void downsample_kernel(const float* __restrict__ x, float* __restrict__ xdn) {
    using namespace csa;
    int idx = blockIdx.x * 256 + threadIdx.x;          // < B*TT*D
    int c = idx % D;
    int t = (idx / D) % TT;
    int b = idx / (D * TT);
    int th = t / 48, tw = t % 48;
    size_t base = ((size_t)b * NT + (size_t)(2 * th) * 96 + 2 * tw) * D + c;
    float v = x[base] + x[base + D] + x[base + 96 * (size_t)D] + x[base + 97 * (size_t)D];
    xdn[idx] = 0.25f * v;
}

// ---------------- per-channel mean of x over (B, N) -> meanx[C] ----------------
__global__ void colmean_kernel(const float* __restrict__ x, float* __restrict__ meanx) {
    using namespace csa;
    __shared__ float red[256];
    int c = blockIdx.x, t = threadIdx.x;
    float s = 0.0f;
    for (int i = t; i < B * NT; i += 256) s += x[(size_t)i * D + c];
    red[t] = s;
    __syncthreads();
    for (int o = 128; o > 0; o >>= 1) { if (t < o) red[t] += red[t + o]; __syncthreads(); }
    if (t == 0) meanx[c] = red[0] * (1.0f / (float)(B * NT));
}

// ---------------- LayerNorm (C=192) -> f16, one wave per row ----------------
__global__ __launch_bounds__(256) void ln_f16_kernel(const float* __restrict__ x,
                                                     const float* __restrict__ g,
                                                     const float* __restrict__ bb,
                                                     _Float16* __restrict__ out, int rows) {
    using namespace csa;
    int wid = threadIdx.x >> 5, lane = threadIdx.x & 31;
    int row = blockIdx.x * 8 + wid;
    if (row >= rows) return;
    const float* xr = x + (size_t)row * D;
    float v[6];
    float s = 0.0f;
#pragma unroll
    for (int i = 0; i < 6; ++i) { v[i] = xr[lane + 32 * i]; s += v[i]; }
#pragma unroll
    for (int m = 16; m; m >>= 1) s += __shfl_xor(s, m, 32);
    float mu = s * (1.0f / (float)D);
    float vs = 0.0f;
#pragma unroll
    for (int i = 0; i < 6; ++i) { float d = v[i] - mu; vs += d * d; }
#pragma unroll
    for (int m = 16; m; m >>= 1) vs += __shfl_xor(vs, m, 32);
    float rstd = rsqrtf(vs * (1.0f / (float)D) + 1e-5f);
    _Float16* orow = out + (size_t)row * D;
#pragma unroll
    for (int i = 0; i < 6; ++i) {
        int c = lane + 32 * i;
        orow[c] = (_Float16)((v[i] - mu) * rstd * g[c] + bb[c]);
    }
}

// ---------------- generic WMMA GEMM: Y[M,N] = A_h[M,K] @ Wt_h[N,K]^T + bias ----
// ep: 0 = f32 out, 1 = f16 out, 2 = exact-GELU f16 out, 3 = resid + f32 out
__global__ __launch_bounds__(128) void gemm16x64_kernel(const _Float16* __restrict__ A,
                                                        const _Float16* __restrict__ Bt,
                                                        const float* __restrict__ bias,
                                                        const float* __restrict__ resid,
                                                        void* __restrict__ outp,
                                                        int M, int N, int K, int ep) {
    int wid = threadIdx.x >> 5, lane = threadIdx.x & 31;
    int w = blockIdx.x * 4 + wid;
    int numM = M >> 4, numN = N >> 6;
    if (w >= numM * numN) return;
    int m0 = (w % numM) << 4;
    int n0 = (w / numM) << 6;
    int l16 = lane & 15, sel = lane >> 4;

    v8f acc[4] = {};
    const _Float16* arow = A + (size_t)(m0 + l16) * K;
    for (int k0 = 0; k0 < K; k0 += 32) {
        half8 alo = *(const half8*)(arow + k0 + sel * 16);
        half8 ahi = *(const half8*)(arow + k0 + sel * 16 + 8);
        v16h av = cat8(alo, ahi);
#pragma unroll
        for (int j = 0; j < 4; ++j) {
            const _Float16* brow = Bt + (size_t)(n0 + j * 16 + l16) * K + k0 + sel * 16;
            v16h bv = *(const v16h*)brow;
            acc[j] = __builtin_amdgcn_wmma_f32_16x16x32_f16(false, av, false, bv,
                                                            (short)0, acc[j], false, false);
        }
    }
#pragma unroll
    for (int j = 0; j < 4; ++j) {
        int col = n0 + j * 16 + l16;
        float bcol = bias[col];
#pragma unroll
        for (int r = 0; r < 8; ++r) {
            int row = m0 + sel * 8 + r;
            float v = acc[j][r] + bcol;
            size_t idx = (size_t)row * N + col;
            if (ep == 0)      ((float*)outp)[idx] = v;
            else if (ep == 1) ((_Float16*)outp)[idx] = (_Float16)v;
            else if (ep == 2) ((_Float16*)outp)[idx] =
                (_Float16)(0.5f * v * (1.0f + erff(v * 0.70710678118654752f)));
            else              ((float*)outp)[idx] = resid[idx] + v;
        }
    }
}

// ---------------- scatter k/v into attention layouts ----------------
// kcat[b][h][key][d]  (B-operand for Q*K^T), vt[b][h][d][key] (B-operand for P*V)
__global__ void scatter_kv_kernel(const _Float16* __restrict__ kh, const _Float16* __restrict__ vh,
                                  _Float16* __restrict__ kcat, _Float16* __restrict__ vt) {
    using namespace csa;
    size_t idx = (size_t)blockIdx.x * 256 + threadIdx.x;   // < B*TT*D
    int c = idx % D;
    size_t bt = idx / D;
    int t = (int)(bt % TT);
    int b = (int)(bt / TT);
    int h = c >> 5, d = c & 31;
    size_t bh = (size_t)b * HEADS + h;
    kcat[(bh * TMP + t) * 32 + d] = kh[idx];
    vt[(bh * 32 + d) * TMP + t]   = vh[idx];
}

__global__ void memrows_kernel(const _Float16* __restrict__ kh, const _Float16* __restrict__ vh,
                               const float* __restrict__ meanx,
                               _Float16* __restrict__ kcat, _Float16* __restrict__ vt) {
    using namespace csa;
    int idx = blockIdx.x * 256 + threadIdx.x;
    if (idx >= MEM * D) return;
    int m = idx / D, c = idx % D;
    float ks = 0.0f, vs = 0.0f;
    for (int b = 0; b < B; ++b)
        for (int j = 0; j < BIN; ++j) {
            size_t r = ((size_t)b * TT + m * BIN + j) * D + c;
            ks += (float)kh[r];
            vs += (float)vh[r];
        }
    float kbar = ks * (1.0f / (float)(B * BIN));
    float vbar = vs * (1.0f / (float)(B * BIN));
    float mk = EMA * meanx[c] + (1.0f - EMA) * kbar;
    float mv = (1.0f - EMA) * vbar;
    int h = c >> 5, d = c & 31;
    for (int b = 0; b < B; ++b) {
        size_t bh = (size_t)b * HEADS + h;
        kcat[(bh * TMP + (TT + m)) * 32 + d] = (_Float16)mk;
        vt[(bh * 32 + d) * TMP + (TT + m)]   = (_Float16)mv;
    }
}

__global__ void pad_kv_kernel(_Float16* __restrict__ kcat, _Float16* __restrict__ vt) {
    using namespace csa;
    int idx = blockIdx.x * 256 + threadIdx.x;   // < B*HEADS*(TMP-TM)*32 = 9216
    int d = idx & 31;
    int p = (idx >> 5) % (TMP - TM);
    int bh = idx / (32 * (TMP - TM));
    int key = TM + p;
    kcat[((size_t)bh * TMP + key) * 32 + d] = (_Float16)0.0f;
    vt[((size_t)bh * 32 + d) * TMP + key]   = (_Float16)0.0f;
}

// ---------------- flash attention: 1 wave per 16-query tile ----------------
__global__ __launch_bounds__(128) void attn_kernel(const _Float16* __restrict__ qh,
                                                   const _Float16* __restrict__ kcat,
                                                   const _Float16* __restrict__ vt,
                                                   _Float16* __restrict__ out) {
    using namespace csa;
    __shared__ __align__(32) _Float16 pbuf[4][16 * 32];
    int wid = threadIdx.x >> 5, lane = threadIdx.x & 31;
    int w = blockIdx.x * 4 + wid;
    const int QT = NT / 16;                       // 576 query tiles
    if (w >= B * HEADS * QT) return;
    int q0 = (w % QT) * 16;
    int h  = (w / QT) % HEADS;
    int b  = w / (QT * HEADS);
    int l16 = lane & 15, sel = lane >> 4;

    // Q tile in A-layout (scale already folded into wq/bq)
    const _Float16* qrow = qh + ((size_t)b * NT + q0 + l16) * D + h * HD;
    v16h qa = cat8(*(const half8*)(qrow + sel * 16), *(const half8*)(qrow + sel * 16 + 8));

    const _Float16* kbase = kcat + ((size_t)b * HEADS + h) * (size_t)TMP * 32;
    const _Float16* vbase = vt   + ((size_t)b * HEADS + h) * (size_t)32 * TMP;

    v8f o0 = {}, o1 = {};
    float m[8], l[8];
#pragma unroll
    for (int r = 0; r < 8; ++r) { m[r] = -1e30f; l[r] = 0.0f; }
    _Float16* pb = pbuf[wid];

    for (int k0 = 0; k0 < TMP; k0 += 32) {
        v16h kb0 = *(const v16h*)(kbase + (size_t)(k0 + l16) * 32 + sel * 16);
        v16h kb1 = *(const v16h*)(kbase + (size_t)(k0 + 16 + l16) * 32 + sel * 16);
        v8f z = {};
        v8f s0 = __builtin_amdgcn_wmma_f32_16x16x32_f16(false, qa, false, kb0, (short)0, z, false, false);
        v8f s1 = __builtin_amdgcn_wmma_f32_16x16x32_f16(false, qa, false, kb1, (short)0, z, false, false);
        // mask padded keys (only final 32-key step)
        if (k0 + l16 >= TM) {
#pragma unroll
            for (int r = 0; r < 8; ++r) s0[r] = -1e30f;
        }
        if (k0 + 16 + l16 >= TM) {
#pragma unroll
            for (int r = 0; r < 8; ++r) s1[r] = -1e30f;
        }
        // online softmax (rows split: lanes 0-15 own rows 0-7, lanes 16-31 rows 8-15)
        float mt[8];
#pragma unroll
        for (int r = 0; r < 8; ++r) mt[r] = fmaxf(s0[r], s1[r]);
#pragma unroll
        for (int msk = 8; msk; msk >>= 1)
#pragma unroll
            for (int r = 0; r < 8; ++r) mt[r] = fmaxf(mt[r], __shfl_xor(mt[r], msk, 16));
        float al[8];
#pragma unroll
        for (int r = 0; r < 8; ++r) {
            float mn = fmaxf(m[r], mt[r]);
            al[r] = __expf(m[r] - mn);
            m[r] = mn;
        }
        float p0[8], p1[8], rs[8];
#pragma unroll
        for (int r = 0; r < 8; ++r) {
            p0[r] = __expf(s0[r] - m[r]);
            p1[r] = __expf(s1[r] - m[r]);
            rs[r] = p0[r] + p1[r];
        }
#pragma unroll
        for (int msk = 8; msk; msk >>= 1)
#pragma unroll
            for (int r = 0; r < 8; ++r) rs[r] += __shfl_xor(rs[r], msk, 16);
#pragma unroll
        for (int r = 0; r < 8; ++r) {
            l[r] = l[r] * al[r] + rs[r];
            o0[r] *= al[r];
            o1[r] *= al[r];
        }
        // C-layout -> A-layout transpose of P through per-wave LDS
#pragma unroll
        for (int r = 0; r < 8; ++r) {
            pb[(sel * 8 + r) * 32 + l16]      = (_Float16)p0[r];
            pb[(sel * 8 + r) * 32 + 16 + l16] = (_Float16)p1[r];
        }
        asm volatile("s_wait_dscnt 0" ::: "memory");
        v16h pa = cat8(*(const half8*)(pb + l16 * 32 + sel * 16),
                       *(const half8*)(pb + l16 * 32 + sel * 16 + 8));
        v16h vb0 = *(const v16h*)(vbase + (size_t)(l16) * TMP + k0 + sel * 16);
        v16h vb1 = *(const v16h*)(vbase + (size_t)(16 + l16) * TMP + k0 + sel * 16);
        o0 = __builtin_amdgcn_wmma_f32_16x16x32_f16(false, pa, false, vb0, (short)0, o0, false, false);
        o1 = __builtin_amdgcn_wmma_f32_16x16x32_f16(false, pa, false, vb1, (short)0, o1, false, false);
    }
#pragma unroll
    for (int r = 0; r < 8; ++r) {
        float inv = 1.0f / l[r];
        size_t row = (size_t)b * NT + q0 + sel * 8 + r;
        out[row * D + h * HD + l16]      = (_Float16)(o0[r] * inv);
        out[row * D + h * HD + 16 + l16] = (_Float16)(o1[r] * inv);
    }
}

// ---------------- host side ----------------
extern "C" void kernel_launch(void* const* d_in, const int* in_sizes, int n_in,
                              void* d_out, int out_size, void* d_ws, size_t ws_size,
                              hipStream_t stream) {
    using namespace csa;
    (void)in_sizes; (void)n_in; (void)out_size; (void)ws_size;

    const float* x     = (const float*)d_in[0];
    const float* lnq_w = (const float*)d_in[1];
    const float* lnq_b = (const float*)d_in[2];
    const float* lnk_w = (const float*)d_in[3];
    const float* lnk_b = (const float*)d_in[4];
    const float* wq = (const float*)d_in[5];
    const float* bq = (const float*)d_in[6];
    const float* wk = (const float*)d_in[7];
    const float* bk = (const float*)d_in[8];
    const float* wv = (const float*)d_in[9];
    const float* bv = (const float*)d_in[10];
    const float* wo = (const float*)d_in[11];
    const float* bo = (const float*)d_in[12];
    const float* ln2_w = (const float*)d_in[13];
    const float* ln2_b = (const float*)d_in[14];
    const float* w1 = (const float*)d_in[15];
    const float* b1 = (const float*)d_in[16];
    const float* w2 = (const float*)d_in[17];
    const float* b2 = (const float*)d_in[18];
    float* out = (float*)d_out;

    char* wsb = (char*)d_ws;
    size_t off = 0;
    auto take = [&](size_t bytes) -> void* {
        void* p = wsb + off;
        off = (off + bytes + 255) & ~(size_t)255;
        return p;
    };
    _Float16* wq_t = (_Float16*)take((size_t)D * D * 2);
    _Float16* wk_t = (_Float16*)take((size_t)D * D * 2);
    _Float16* wv_t = (_Float16*)take((size_t)D * D * 2);
    _Float16* wo_t = (_Float16*)take((size_t)D * D * 2);
    _Float16* w1_t = (_Float16*)take((size_t)D * MLPH * 2);
    _Float16* w2_t = (_Float16*)take((size_t)MLPH * D * 2);
    float* bq_s  = (float*)take(D * 4);
    float* meanx = (float*)take(D * 4);
    float* x_dn  = (float*)take((size_t)B * TT * D * 4);
    _Float16* xq_h  = (_Float16*)take((size_t)B * NT * D * 2);
    _Float16* xkv_h = (_Float16*)take((size_t)B * TT * D * 2);
    _Float16* q_h   = (_Float16*)take((size_t)B * NT * D * 2);
    _Float16* k_h   = (_Float16*)take((size_t)B * TT * D * 2);
    _Float16* v_h   = (_Float16*)take((size_t)B * TT * D * 2);
    _Float16* kcat  = (_Float16*)take((size_t)B * HEADS * TMP * 32 * 2);
    _Float16* vt    = (_Float16*)take((size_t)B * HEADS * 32 * TMP * 2);
    _Float16* attn_h = (_Float16*)take((size_t)B * NT * D * 2);
    float* out1  = (float*)take((size_t)B * NT * D * 4);
    _Float16* h2_h = (_Float16*)take((size_t)B * NT * D * 2);
    _Float16* g_h  = (_Float16*)take((size_t)B * NT * MLPH * 2);

    // 1) weight prep (scale folded into wq / bq)
    wprep_kernel<<<(D * D + 255) / 256, 256, 0, stream>>>(wq, wq_t, D, D, SCALE);
    wprep_kernel<<<(D * D + 255) / 256, 256, 0, stream>>>(wk, wk_t, D, D, 1.0f);
    wprep_kernel<<<(D * D + 255) / 256, 256, 0, stream>>>(wv, wv_t, D, D, 1.0f);
    wprep_kernel<<<(D * D + 255) / 256, 256, 0, stream>>>(wo, wo_t, D, D, 1.0f);
    wprep_kernel<<<(D * MLPH + 255) / 256, 256, 0, stream>>>(w1, w1_t, D, MLPH, 1.0f);
    wprep_kernel<<<(MLPH * D + 255) / 256, 256, 0, stream>>>(w2, w2_t, MLPH, D, 1.0f);
    sbias_kernel<<<1, 256, 0, stream>>>(bq, bq_s, SCALE);

    // 2) downsample + memory-init column mean
    downsample_kernel<<<(B * TT * D) / 256, 256, 0, stream>>>(x, x_dn);
    colmean_kernel<<<D, 256, 0, stream>>>(x, meanx);

    // 3) LayerNorms -> f16
    ln_f16_kernel<<<(B * NT + 7) / 8, 256, 0, stream>>>(x, lnq_w, lnq_b, xq_h, B * NT);
    ln_f16_kernel<<<(B * TT + 7) / 8, 256, 0, stream>>>(x_dn, lnk_w, lnk_b, xkv_h, B * TT);

    // 4) projections (WMMA)
    {
        int wq_waves = (B * NT / 16) * (D / 64);
        gemm16x64_kernel<<<wq_waves / 4, 128, 0, stream>>>(xq_h, wq_t, bq_s, nullptr, q_h,
                                                           B * NT, D, D, 1);
        int kv_waves = (B * TT / 16) * (D / 64);
        gemm16x64_kernel<<<kv_waves / 4, 128, 0, stream>>>(xkv_h, wk_t, bk, nullptr, k_h,
                                                           B * TT, D, D, 1);
        gemm16x64_kernel<<<kv_waves / 4, 128, 0, stream>>>(xkv_h, wv_t, bv, nullptr, v_h,
                                                           B * TT, D, D, 1);
    }

    // 5) K/V layouts + EMA memory rows + zero pad
    scatter_kv_kernel<<<(B * TT * D) / 256, 256, 0, stream>>>(k_h, v_h, kcat, vt);
    memrows_kernel<<<(MEM * D + 255) / 256, 256, 0, stream>>>(k_h, v_h, meanx, kcat, vt);
    pad_kv_kernel<<<(B * HEADS * (TMP - TM) * 32) / 256, 256, 0, stream>>>(kcat, vt);

    // 6) flash attention (WMMA)
    {
        int tiles = B * HEADS * (NT / 16);   // 6912
        attn_kernel<<<tiles / 4, 128, 0, stream>>>(q_h, kcat, vt, attn_h);
    }

    // 7) output projection -> f32
    {
        int waves = (B * NT / 16) * (D / 64);
        gemm16x64_kernel<<<waves / 4, 128, 0, stream>>>(attn_h, wo_t, bo, nullptr, out1,
                                                        B * NT, D, D, 0);
    }

    // 8) MLP with residual
    ln_f16_kernel<<<(B * NT + 7) / 8, 256, 0, stream>>>(out1, ln2_w, ln2_b, h2_h, B * NT);
    {
        int w1_waves = (B * NT / 16) * (MLPH / 64);
        gemm16x64_kernel<<<w1_waves / 4, 128, 0, stream>>>(h2_h, w1_t, b1, nullptr, g_h,
                                                           B * NT, MLPH, D, 2);
        int w2_waves = (B * NT / 16) * (D / 64);
        gemm16x64_kernel<<<w2_waves / 4, 128, 0, stream>>>(g_h, w2_t, b2, out1, out,
                                                           B * NT, D, MLPH, 3);
    }
}
